// UV_Aggregator_NoUserAttention_29669634080987
// MI455X (gfx1250) — matrix operand order repaired and checked by reading, not compiled, not run on hardware
//
#include <hip/hip_runtime.h>

typedef __attribute__((ext_vector_type(16))) _Float16 v16h;
typedef __attribute__((ext_vector_type(8)))  _Float16 v8h;
typedef __attribute__((ext_vector_type(8)))  float    v8f;

#define NB      4096
#define LHIST   200
#define NTILES  13   // ceil(200/16); last tile has 8 valid rows

// ---- A-fragment builders -------------------------------------------------
// Per ISA 7.12.2 (16-bit A 16x32): lanes 0-15 hold row M=lane, K {0..7,16..23};
// lanes 16-31 hold row M=lane-16, K {8..15,24..31}. halfs[0..7]=chunk0,
// halfs[8..15]=chunk1.

// f32 source (fallback path): load 2x8 floats, convert.
__device__ __forceinline__ v16h load_a16(const float* __restrict__ p0,
                                         const float* __restrict__ p1) {
  float4 f0 = *(const float4*)(p0);
  float4 f1 = *(const float4*)(p0 + 4);
  float4 g0 = *(const float4*)(p1);
  float4 g1 = *(const float4*)(p1 + 4);
  v16h a;
  a[0]=(_Float16)f0.x;  a[1]=(_Float16)f0.y;  a[2]=(_Float16)f0.z;  a[3]=(_Float16)f0.w;
  a[4]=(_Float16)f1.x;  a[5]=(_Float16)f1.y;  a[6]=(_Float16)f1.z;  a[7]=(_Float16)f1.w;
  a[8]=(_Float16)g0.x;  a[9]=(_Float16)g0.y;  a[10]=(_Float16)g0.z; a[11]=(_Float16)g0.w;
  a[12]=(_Float16)g1.x; a[13]=(_Float16)g1.y; a[14]=(_Float16)g1.z; a[15]=(_Float16)g1.w;
  return a;
}

// f16 source (fast path): two 16-byte loads, no conversion VALU.
__device__ __forceinline__ v16h load_a16(const _Float16* __restrict__ p0,
                                         const _Float16* __restrict__ p1) {
  v8h lo = *(const v8h*)(p0);
  v8h up = *(const v8h*)(p1);
  return __builtin_shufflevector(lo, up, 0, 1, 2, 3, 4, 5, 6, 7,
                                         8, 9, 10, 11, 12, 13, 14, 15);
}

// Select pointer into the concatenated x = [e_uv | e_r]; 8-wide chunks never
// straddle the 64 boundary.
template <typename ET>
__device__ __forceinline__ const ET* xptr(const ET* __restrict__ uv,
                                          const ET* __restrict__ rr, int k) {
  return (k < 64) ? (uv + k) : (rr + (k - 64));
}

// ---- one-time f32 -> f16 table conversion into workspace ----------------
__global__ void cvt_f32_to_f16_kernel(const float* __restrict__ src,
                                      _Float16* __restrict__ dst, int n) {
  for (int i = blockIdx.x * blockDim.x + threadIdx.x; i < n;
       i += gridDim.x * blockDim.x)
    dst[i] = (_Float16)src[i];
}

// ---- main fused kernel (ET = _Float16 fast path, float fallback) ---------
template <typename ET>
__global__ __launch_bounds__(256)
void uvagg_wmma_kernel(const ET*    __restrict__ v2e,   // [100000,64]
                       const ET*    __restrict__ r2e,   // [5,64]
                       const float* __restrict__ w1W,   // [64,128] (o,i)
                       const float* __restrict__ w1b,   // [64]
                       const float* __restrict__ w2W,   // [64,64]  (o,i)
                       const float* __restrict__ w2b,   // [64]
                       const int*   __restrict__ huv,   // [4096,200]
                       const int*   __restrict__ hr,    // [4096,200]
                       float*       __restrict__ out)   // [4096,64]
{
  __shared__ _Float16 sW1[64 * 128];   // 16 KB f16 W1, B-fragment friendly [n][k]
  __shared__ _Float16 sW2[64 * 64];    // 8 KB
  __shared__ _Float16 sX[8][16 * 64];  // 16 KB per-wave h staging (D -> A layout)
  __shared__ float    sAcc[64];

  const int tid     = threadIdx.x;
  const int wave    = tid >> 5;        // wave32
  const int lane    = tid & 31;
  const int laneLow = lane & 15;
  const int hi      = lane >> 4;

  for (int i = tid; i < 64 * 128; i += 256) sW1[i] = (_Float16)w1W[i];
  for (int i = tid; i < 64 * 64;  i += 256) sW2[i] = (_Float16)w2W[i];

  float bias1[4], bias2[4];
#pragma unroll
  for (int nt = 0; nt < 4; ++nt) {
    bias1[nt] = w1b[nt * 16 + laneLow];
    bias2[nt] = w2b[nt * 16 + laneLow];
  }

  _Float16* __restrict__ xs = &sX[wave][0];

  for (int b = blockIdx.x; b < NB; b += gridDim.x) {
    if (tid < 64) sAcc[tid] = 0.0f;
    __syncthreads();   // also guards first use of sW1/sW2

    float acc[4] = {0.f, 0.f, 0.f, 0.f};

    for (int tile = wave; tile < NTILES; tile += 8) {
      const bool fullTile = (tile * 16 + 16) <= LHIST;   // wave-uniform

      // ---- gather the 16 token rows of this tile ----
      const int tok  = tile * 16 + laneLow;
      const int tokc = (tok < LHIST) ? tok : (LHIST - 1);
      const int iuv  = huv[(size_t)b * LHIST + tokc];
      const int ir   = hr [(size_t)b * LHIST + tokc];
      const ET* __restrict__ rowUV = v2e + (size_t)iuv * 64;
      const ET* __restrict__ rowR  = r2e + (size_t)ir  * 64;

      // ---- A fragments for GEMM1 (x: 16x128 -> 4 K-steps of 16x32) ----
      v16h a1[4];
#pragma unroll
      for (int ks = 0; ks < 4; ++ks) {
        const int c0 = ks * 32 + hi * 8;
        a1[ks] = load_a16(xptr(rowUV, rowR, c0), xptr(rowUV, rowR, c0 + 16));
      }

      // ---- GEMM1: h = relu(x*W1^T + b1); stage h (f16) for transpose ----
#pragma unroll
      for (int nt = 0; nt < 4; ++nt) {
        const int ncol = nt * 16 + laneLow;
        v8f c;
#pragma unroll
        for (int i = 0; i < 8; ++i) c[i] = bias1[nt];
#pragma unroll
        for (int ks = 0; ks < 4; ++ks) {
          v16h bf = __builtin_shufflevector(*(const v8h*)(sW1 + ncol * 128 + ks * 32 + hi * 8), *(const v8h*)(sW1 + ncol * 128 + ks * 32 + 16 + hi * 8), 0, 1, 2, 3, 4, 5, 6, 7, 8, 9, 10, 11, 12, 13, 14, 15);
          c = __builtin_amdgcn_wmma_f32_16x16x32_f16(false, a1[ks], false, bf,
                                                     (short)0, c, false, false);
        }
#pragma unroll
        for (int r = 0; r < 8; ++r)
          xs[(hi * 8 + r) * 64 + ncol] = (_Float16)fmaxf(c[r], 0.0f);
      }
      // Same-wave DS is in-order; make the stores visible to the reloads.
      asm volatile("s_wait_dscnt 0" ::: "memory");

      // ---- reload h in A layout (16x64 -> 2 K-steps) ----
      const _Float16* __restrict__ xrow = xs + laneLow * 64;
      v16h a2[2];
#pragma unroll
      for (int ks = 0; ks < 2; ++ks) {
        const int c0 = ks * 32 + hi * 8;
        a2[ks] = load_a16(xrow + c0, xrow + c0 + 16);
      }

      // ---- GEMM2: o = relu(h*W2^T + b2); row-sum into per-lane acc ----
#pragma unroll
      for (int nt = 0; nt < 4; ++nt) {
        const int ncol = nt * 16 + laneLow;
        v8f c;
#pragma unroll
        for (int i = 0; i < 8; ++i) c[i] = bias2[nt];
#pragma unroll
        for (int ks = 0; ks < 2; ++ks) {
          v16h bf = __builtin_shufflevector(*(const v8h*)(sW2 + ncol * 64 + ks * 32 + hi * 8), *(const v8h*)(sW2 + ncol * 64 + ks * 32 + 16 + hi * 8), 0, 1, 2, 3, 4, 5, 6, 7, 8, 9, 10, 11, 12, 13, 14, 15);
          c = __builtin_amdgcn_wmma_f32_16x16x32_f16(false, a2[ks], false, bf,
                                                     (short)0, c, false, false);
        }
        float s = 0.0f;
        if (fullTile) {
#pragma unroll
          for (int r = 0; r < 8; ++r) s += fmaxf(c[r], 0.0f);
        } else {
#pragma unroll
          for (int r = 0; r < 8; ++r) {
            const int tokr = tile * 16 + hi * 8 + r;
            s += (tokr < LHIST) ? fmaxf(c[r], 0.0f) : 0.0f;
          }
        }
        acc[nt] += s;
      }
    } // tile loop

    // Merge lane halves (lane n / n+16: same column, disjoint rows).
#pragma unroll
    for (int nt = 0; nt < 4; ++nt) acc[nt] += __shfl_xor(acc[nt], 16, 32);
    if (hi == 0) {
#pragma unroll
      for (int nt = 0; nt < 4; ++nt) atomicAdd(&sAcc[nt * 16 + laneLow], acc[nt]);
    }
    __syncthreads();
    if (tid < 64) out[(size_t)b * 64 + tid] = sAcc[tid] * (1.0f / (float)LHIST);
    __syncthreads();   // protect sAcc before next iteration re-zeroes it
  }
}

extern "C" void kernel_launch(void* const* d_in, const int* in_sizes, int n_in,
                              void* d_out, int out_size, void* d_ws, size_t ws_size,
                              hipStream_t stream) {
  (void)n_in; (void)out_size;
  const float* v2e = (const float*)d_in[0];
  const float* r2e = (const float*)d_in[1];
  const float* w1W = (const float*)d_in[2];
  const float* w1b = (const float*)d_in[3];
  const float* w2W = (const float*)d_in[4];
  const float* w2b = (const float*)d_in[5];
  // d_in[6] = nodes (unused in the uv=True branch)
  const int* huv = (const int*)d_in[7];
  const int* hr  = (const int*)d_in[8];
  float* out = (float*)d_out;

  const int nV = in_sizes[0];               // 100000*64
  const int nR = in_sizes[1];               // 5*64
  const size_t offR   = ((size_t)nV * 2 + 255) & ~(size_t)255;
  const size_t needWs = offR + (size_t)nR * 2;

  dim3 grid(1024), block(256);

  if (d_ws != nullptr && ws_size >= needWs) {
    // Fast path: pre-convert embedding tables to f16 in workspace (halves the
    // random-gather bytes and removes all per-tile cvt VALU).
    _Float16* v2eH = (_Float16*)d_ws;
    _Float16* r2eH = (_Float16*)((char*)d_ws + offR);
    hipLaunchKernelGGL(cvt_f32_to_f16_kernel, dim3(2048), dim3(256), 0, stream,
                       v2e, v2eH, nV);
    hipLaunchKernelGGL(cvt_f32_to_f16_kernel, dim3(1), dim3(256), 0, stream,
                       r2e, r2eH, nR);
    hipLaunchKernelGGL(uvagg_wmma_kernel<_Float16>, grid, block, 0, stream,
                       (const _Float16*)v2eH, (const _Float16*)r2eH,
                       w1W, w1b, w2W, w2b, huv, hr, out);
  } else {
    // Fallback: gather f32 rows and convert in-register.
    hipLaunchKernelGGL(uvagg_wmma_kernel<float>, grid, block, 0, stream,
                       v2e, r2e, w1W, w1b, w2W, w2b, huv, hr, out);
  }
}
